// W8A8PDMixLinear_47356309405794
// MI455X (gfx1250) — 8-bit matrix operand order repaired and checked by reading, compiled, not run on hardware
//
#include <hip/hip_runtime.h>
#include <stdint.h>

typedef __attribute__((ext_vector_type(8))) int v8i;

#define MSZ 4096
#define KSZ 4096
#define NSZ 4096
#define BM 128
#define BN 256
#define BK 64

__device__ __forceinline__ int quant1(float v, float inv, float o) {
    float r = rintf(fmaf(v, inv, o));
    r = fminf(127.0f, fmaxf(-128.0f, r));
    return (int)r;
}

__global__ __launch_bounds__(256) void quant_x_kernel(
    const float* __restrict__ x,
    const float* __restrict__ input_scale,
    const int* __restrict__ input_offset,
    uint32_t* __restrict__ xq) {
    size_t i = (size_t)blockIdx.x * blockDim.x + threadIdx.x;
    const float inv = 1.0f / input_scale[0];
    const float o = (float)input_offset[0];
    const float4 v = reinterpret_cast<const float4*>(x)[i];
    int q0 = quant1(v.x, inv, o);
    int q1 = quant1(v.y, inv, o);
    int q2 = quant1(v.z, inv, o);
    int q3 = quant1(v.w, inv, o);
    xq[i] = (uint32_t)(q0 & 0xff) | ((uint32_t)(q1 & 0xff) << 8) |
            ((uint32_t)(q2 & 0xff) << 16) | ((uint32_t)(q3 & 0xff) << 24);
}

__global__ __launch_bounds__(256) void pack_w_kernel(
    const int* __restrict__ w, uint32_t* __restrict__ wq) {
    size_t i = (size_t)blockIdx.x * blockDim.x + threadIdx.x;
    const int4 v = reinterpret_cast<const int4*>(w)[i];
    wq[i] = (uint32_t)(v.x & 0xff) | ((uint32_t)(v.y & 0xff) << 8) |
            ((uint32_t)(v.z & 0xff) << 16) | ((uint32_t)(v.w & 0xff) << 24);
}

__global__ __launch_bounds__(256, 1) void w8a8_wmma_kernel(
    const uint8_t* __restrict__ xq,
    const uint8_t* __restrict__ wq,
    const int* __restrict__ quant_bias,
    const float* __restrict__ deq_scale,
    float* __restrict__ out) {
    __shared__ __align__(16) uint8_t As[2][BM * BK];
    __shared__ __align__(16) uint8_t Bs[2][BN * BK];

    const int t = threadIdx.x;
    const int lane = t & 31;
    const int wave = t >> 5;
    const int waveM = wave & 1;
    const int waveN = wave >> 1;
    const int h = lane >> 4;
    const int l16 = lane & 15;

    const int mbase = blockIdx.y * BM;
    const int nbase = blockIdx.x * BN;

    const uint8_t* gA[2];
    const uint8_t* gB[4];
#pragma unroll
    for (int i = 0; i < 2; ++i) {
        int e = t * 16 + i * 4096;
        gA[i] = xq + (size_t)(mbase + (e >> 6)) * KSZ + (e & 63);
    }
#pragma unroll
    for (int i = 0; i < 4; ++i) {
        int e = t * 16 + i * 4096;
        gB[i] = wq + (size_t)(nbase + (e >> 6)) * KSZ + (e & 63);
    }
    const uint32_t ldsA0 = (uint32_t)(uintptr_t)(&As[0][0]);
    const uint32_t ldsB0 = (uint32_t)(uintptr_t)(&Bs[0][0]);

    auto stage = [&](int b, int kt) {
#pragma unroll
        for (int i = 0; i < 2; ++i) {
            uint32_t la = ldsA0 + (uint32_t)(b * (BM * BK) + t * 16 + i * 4096);
            uint64_t ga = (uint64_t)(uintptr_t)(gA[i] + (size_t)kt * BK);
            asm volatile("global_load_async_to_lds_b128 %0, %1, off"
                         :: "v"(la), "v"(ga) : "memory");
        }
#pragma unroll
        for (int i = 0; i < 4; ++i) {
            uint32_t lb = ldsB0 + (uint32_t)(b * (BN * BK) + t * 16 + i * 4096);
            uint64_t gb = (uint64_t)(uintptr_t)(gB[i] + (size_t)kt * BK);
            asm volatile("global_load_async_to_lds_b128 %0, %1, off"
                         :: "v"(lb), "v"(gb) : "memory");
        }
    };

    int ncol[4];
    int qb[4];
    float dsc[4];
#pragma unroll
    for (int tj = 0; tj < 4; ++tj) {
        ncol[tj] = nbase + (waveN * 4 + tj) * 16 + l16;
        qb[tj] = quant_bias[ncol[tj]];
        dsc[tj] = deq_scale[ncol[tj]];
    }

    v8i acc[4][4] = {};

    stage(0, 0);
    asm volatile("s_wait_asynccnt 0" ::: "memory");
    __syncthreads();

    const int KT = KSZ / BK;
    int buf = 0;
    for (int kt = 0; kt < KT; ++kt) {
        if (kt + 1 < KT) stage(buf ^ 1, kt + 1);

        const uint8_t* Ab = &As[buf][0];
        const uint8_t* Bb = &Bs[buf][0];

        v8i af[4];
#pragma unroll
        for (int ti = 0; ti < 4; ++ti) {
            int base = (waveM * 64 + ti * 16 + l16) * BK + h * 16;
            int2 q0 = *reinterpret_cast<const int2*>(Ab + base + 0);
            int2 q1 = *reinterpret_cast<const int2*>(Ab + base + 8);
            int2 q2 = *reinterpret_cast<const int2*>(Ab + base + 32);
            int2 q3 = *reinterpret_cast<const int2*>(Ab + base + 40);
            af[ti][0] = q0.x; af[ti][1] = q0.y;
            af[ti][2] = q1.x; af[ti][3] = q1.y;
            af[ti][4] = q2.x; af[ti][5] = q2.y;
            af[ti][6] = q3.x; af[ti][7] = q3.y;
        }

#pragma unroll
        for (int tj = 0; tj < 4; ++tj) {
            int base = (waveN * 64 + tj * 16 + l16) * BK + h * 16;
            int4 p0 = *reinterpret_cast<const int4*>(Bb + base + 0);
            int4 p1 = *reinterpret_cast<const int4*>(Bb + base + 32);
            v8i bf;
            bf[0] = p0.x; bf[1] = p0.y; bf[2] = p0.z; bf[3] = p0.w;
            bf[4] = p1.x; bf[5] = p1.y; bf[6] = p1.z; bf[7] = p1.w;
#pragma unroll
            for (int ti = 0; ti < 4; ++ti)
                acc[ti][tj] = __builtin_amdgcn_wmma_i32_16x16x64_iu8(
                     true, af[ti],  true, bf,
                    acc[ti][tj],  false,  false);
        }

        asm volatile("s_wait_asynccnt 0" ::: "memory");
        __syncthreads();
        buf ^= 1;
    }

#pragma unroll
    for (int ti = 0; ti < 4; ++ti) {
        int mrow0 = mbase + (waveM * 4 + ti) * 16 + h * 8;
#pragma unroll
        for (int tj = 0; tj < 4; ++tj) {
#pragma unroll
            for (int r = 0; r < 8; ++r) {
                size_t idx = (size_t)(mrow0 + r) * NSZ + ncol[tj];
                out[idx] = (float)(acc[ti][tj][r] + qb[tj]) * dsc[tj];
            }
        }
    }
}

extern "C" void kernel_launch(void* const* d_in, const int* in_sizes, int n_in,
                              void* d_out, int out_size, void* d_ws, size_t ws_size,
                              hipStream_t stream) {
    const float* x            = (const float*)d_in[0];
    const int*   weight       = (const int*)d_in[1];
    const float* deq_scale    = (const float*)d_in[2];
    const int*   quant_bias   = (const int*)d_in[3];
    const float* input_scale  = (const float*)d_in[4];
    const int*   input_offset = (const int*)d_in[5];
    float* out = (float*)d_out;

    uint8_t* xq = (uint8_t*)d_ws;
    uint8_t* wq = xq + (size_t)MSZ * KSZ;

    quant_x_kernel<<<(MSZ * (size_t)KSZ / 4) / 256, 256, 0, stream>>>(
        x, input_scale, input_offset, (uint32_t*)xq);
    pack_w_kernel<<<(NSZ * (size_t)KSZ / 4) / 256, 256, 0, stream>>>(
        weight, (uint32_t*)wq);

    dim3 grid(NSZ / BN, MSZ / BM);
    w8a8_wmma_kernel<<<grid, 256, 0, stream>>>(xq, wq, quant_bias, deq_scale, out);
}
